// EquivariantTPConv_65489661329903
// MI455X (gfx1250) — matrix operand order repaired and checked by reading, not compiled, not run on hardware
//
#include <hip/hip_runtime.h>
#include <hip/hip_bf16.h>

typedef __attribute__((ext_vector_type(16))) _Float16 v16h;
typedef __attribute__((ext_vector_type(8)))  _Float16 v8h;
typedef __attribute__((ext_vector_type(8)))  float    v8f;
typedef __attribute__((ext_vector_type(4)))  float    v4f;

#define N_PATHS 3
#define MSRC 64
#define MDST 16
#define ESD 32
#define HID 128
#define WNUM (N_PATHS*MSRC*MDST)   // 3072
#define OUTC 144
#define ETILE 32                   // edges per wave (2 x 16-row WMMA sets)

__global__ void zero_kernel(float* __restrict__ out, int out_size,
                            float* __restrict__ cnt, int n_dst) {
  int i = blockIdx.x * blockDim.x + threadIdx.x;
  if (i < out_size) out[i] = 0.0f;
  if (i < n_dst)    cnt[i] = 0.0f;
}

// W1 [32][128] -> W1T [128 cols][32 k] f16 ; W2 [128][3072] -> W2T [3072 cols][128 h] f16
__global__ void prep_weights(const float* __restrict__ W1, const float* __restrict__ W2,
                             _Float16* __restrict__ W1T, _Float16* __restrict__ W2T) {
  int i = blockIdx.x * blockDim.x + threadIdx.x;
  if (i < HID*ESD) {
    int col = i / ESD, k = i % ESD;
    W1T[i] = (_Float16)W1[k*HID + col];
  }
  int j = i - HID*ESD;
  if (j >= 0 && j < WNUM*HID) {
    int col = j / HID, h = j % HID;
    W2T[j] = (_Float16)W2[(size_t)h*WNUM + col];
  }
}

__device__ __forceinline__ float silu_fast(float v) {
  // v * sigmoid(v) with v_rcp_f32 instead of IEEE divide
  float t = 1.0f + __expf(-v);
  return v * __builtin_amdgcn_rcpf(t);
}

__global__ void __launch_bounds__(32)
edge_kernel(const float* __restrict__ h_src,
            const float* __restrict__ edge_vec,
            const float* __restrict__ edge_scalars,
            const float* __restrict__ b1,
            const float* __restrict__ b2,
            const int*   __restrict__ src_idx,
            const int*   __restrict__ dst_idx,
            const _Float16* __restrict__ W1T,
            const _Float16* __restrict__ W2T,
            float* __restrict__ out,
            float* __restrict__ cnt,
            int E)
{
  __shared__ alignas(16) _Float16 esF16[ETILE*ESD];   // [e][c]
  __shared__ alignas(16) _Float16 hidF16[ETILE*HID];  // [e][h]
  __shared__ alignas(16) float    xT[MSRC*ETILE];     // [u][e]
  __shared__ float shL[ETILE*9];
  __shared__ int   dstL[ETILE];

  const int lane = threadIdx.x;
  const int Mn   = lane & 15;   // WMMA column (v index)
  const int hi   = lane >> 4;   // lane half
  const int e0   = blockIdx.x * ETILE;

  // ---- stage dst ids, counts, spherical harmonics (one edge per lane) ----
  {
    int e  = e0 + lane;
    int ec = e < E ? e : E - 1;
    int d  = dst_idx[ec];
    dstL[lane] = d;
    if (e < E) atomicAdd(&cnt[d], 1.0f);
    float vx = edge_vec[ec*3+0], vy = edge_vec[ec*3+1], vz = edge_vec[ec*3+2];
    float nrm = sqrtf(vx*vx + vy*vy + vz*vz);
    float inv = 1.0f / fmaxf(nrm, 1e-9f);
    float x = vx*inv, y = vy*inv, z = vz*inv;
    const float SQ3  = 1.7320508075688772f;
    const float SQ5  = 2.23606797749979f;
    const float SQ15 = 3.872983346207417f;
    shL[lane*9+0] = 1.0f;
    shL[lane*9+1] = SQ3*x;
    shL[lane*9+2] = SQ3*y;
    shL[lane*9+3] = SQ3*z;
    shL[lane*9+4] = SQ15*x*y;
    shL[lane*9+5] = SQ15*y*z;
    shL[lane*9+6] = 0.5f*SQ5*(3.0f*z*z - 1.0f);
    shL[lane*9+7] = SQ15*x*z;
    shL[lane*9+8] = 0.5f*SQ15*(x*x - y*y);
  }

  // ---- stage edge_scalars as f16: one edge (32 scalars) per lane ----
  {
    int e = e0 + lane; int ec = e < E ? e : E - 1;
    const float* src = edge_scalars + (size_t)ec*ESD;
    #pragma unroll
    for (int k = 0; k < 8; ++k) {
      v4f v = *(const v4f*)(src + 4*k);
      #pragma unroll
      for (int t = 0; t < 4; ++t)
        esF16[lane*ESD + 4*k + t] = (_Float16)v[t];
    }
  }

  // ---- stage gathered x = h_src[src_idx], transposed [u][e]: one edge per lane ----
  {
    int e = e0 + lane; int ec = e < E ? e : E - 1;
    int row = src_idx[ec];
    const float* src = h_src + (size_t)row*MSRC;
    #pragma unroll
    for (int k = 0; k < 16; ++k) {
      v4f v = *(const v4f*)(src + 4*k);
      #pragma unroll
      for (int t = 0; t < 4; ++t)
        xT[(4*k + t)*ETILE + lane] = v[t];
    }
  }

  __syncthreads();

  // ---- phase 1: hidden = silu(es @ W1 + b1) via WMMA, two 16-edge sets ----
  #pragma unroll
  for (int s = 0; s < 2; ++s) {
    v16h a_es;
    const _Float16* arow = esF16 + (16*s + Mn)*ESD + 16*hi;
    *((v8h*)&a_es)     = *(const v8h*)(arow);
    *((v8h*)&a_es + 1) = *(const v8h*)(arow + 8);
    #pragma unroll
    for (int t = 0; t < 8; ++t) {
      int n0 = t * 16;
      float bv = b1[n0 + Mn];
      v8f c;
      #pragma unroll
      for (int j = 0; j < 8; ++j) c[j] = bv;
      v16h bw = *(const v16h*)(W1T + (n0 + Mn)*ESD + 16*hi);
      c = __builtin_amdgcn_wmma_f32_16x16x32_f16(false, a_es, false, bw,
                                                 (short)0, c, false, false);
      #pragma unroll
      for (int j = 0; j < 8; ++j)
        hidF16[(16*s + j + 8*hi)*HID + n0 + Mn] = (_Float16)silu_fast(c[j]);
    }
  }

  __syncthreads();

  // ---- load A tiles of hidden (K = 128 -> 4 tiles) for both edge sets ----
  v16h a[2][4];
  #pragma unroll
  for (int s = 0; s < 2; ++s) {
    const _Float16* arow = hidF16 + (16*s + Mn)*HID + 16*hi;
    #pragma unroll
    for (int ks = 0; ks < 4; ++ks) {
      *((v8h*)&a[s][ks])     = *(const v8h*)(arow + ks*32);
      *((v8h*)&a[s][ks] + 1) = *(const v8h*)(arow + ks*32 + 8);
    }
  }

  // ---- main loop: wtile(l,u) = hidden @ W2[:, l,u,:] + b2 ; feat += x[:,u] * wtile
  //      each B tile is loaded once and consumed by BOTH 16-edge sets (2x reuse)
  const float scale = 0.125f;   // 1/sqrt(MUL_SRC)
  #pragma unroll
  for (int l = 0; l < N_PATHS; ++l) {
    v8f facc[2];
    #pragma unroll
    for (int s = 0; s < 2; ++s)
      #pragma unroll
      for (int j = 0; j < 8; ++j) facc[s][j] = 0.0f;

    for (int u = 0; u < MSRC; ++u) {
      int n0 = l*(MSRC*MDST) + u*MDST;
      float bv = b2[n0 + Mn];                       // L2-hot, coalesced
      const _Float16* w2row = W2T + (size_t)(n0 + Mn)*HID + 16*hi;
      if (u + 2 < MSRC)                             // prefetch 2 B-tiles ahead
        __builtin_prefetch(w2row + 2*MDST*HID, 0, 1);
      v8f c0, c1;
      #pragma unroll
      for (int j = 0; j < 8; ++j) { c0[j] = bv; c1[j] = bv; }
      #pragma unroll
      for (int ks = 0; ks < 4; ++ks) {
        v16h b = *(const v16h*)(w2row + ks*32);
        c0 = __builtin_amdgcn_wmma_f32_16x16x32_f16(false, a[0][ks], false, b,
                                                    (short)0, c0, false, false);
        c1 = __builtin_amdgcn_wmma_f32_16x16x32_f16(false, a[1][ks], false, b,
                                                    (short)0, c1, false, false);
      }
      #pragma unroll
      for (int s = 0; s < 2; ++s) {
        v4f xv0 = *(const v4f*)&xT[u*ETILE + 16*s + 8*hi];
        v4f xv1 = *(const v4f*)&xT[u*ETILE + 16*s + 8*hi + 4];
        const v8f& c = s ? c1 : c0;
        #pragma unroll
        for (int j = 0; j < 4; ++j) {
          facc[s][j]     += xv0[j] * c[j];
          facc[s][j + 4] += xv1[j] * c[j + 4];
        }
      }
    }

    // scatter: msg = feat * sh, atomically accumulated per destination
    #pragma unroll
    for (int s = 0; s < 2; ++s) {
      #pragma unroll
      for (int j = 0; j < 8; ++j) {
        int eL = 16*s + j + 8*hi;
        if (e0 + eL < E) {
          float f = facc[s][j] * scale;
          float* orow = out + (size_t)dstL[eL]*OUTC;
          if (l == 0) {
            atomicAdd(orow + Mn, f * shL[eL*9 + 0]);
          } else if (l == 1) {
            #pragma unroll
            for (int m = 0; m < 3; ++m)
              atomicAdd(orow + 16 + Mn*3 + m, f * shL[eL*9 + 1 + m]);
          } else {
            #pragma unroll
            for (int m = 0; m < 5; ++m)
              atomicAdd(orow + 64 + Mn*5 + m, f * shL[eL*9 + 4 + m]);
          }
        }
      }
    }
  }
}

__global__ void finalize_kernel(float* __restrict__ out,
                                const float* __restrict__ cnt, int out_size) {
  int i = blockIdx.x * blockDim.x + threadIdx.x;
  if (i < out_size) {
    int d = i / OUTC;
    out[i] /= fmaxf(cnt[d], 1.0f);
  }
}

extern "C" void kernel_launch(void* const* d_in, const int* in_sizes, int n_in,
                              void* d_out, int out_size, void* d_ws, size_t ws_size,
                              hipStream_t stream) {
  (void)n_in; (void)ws_size;
  const float* h_src        = (const float*)d_in[0];
  const float* edge_vec     = (const float*)d_in[1];
  const float* edge_scalars = (const float*)d_in[2];
  const float* W1           = (const float*)d_in[3];
  const float* b1           = (const float*)d_in[4];
  const float* W2           = (const float*)d_in[5];
  const float* b2           = (const float*)d_in[6];
  const int*   src_idx      = (const int*)d_in[7];
  const int*   dst_idx      = (const int*)d_in[8];

  int E     = in_sizes[1] / 3;
  int n_dst = out_size / OUTC;
  float* out = (float*)d_out;

  char* ws = (char*)d_ws;
  float* cnt = (float*)ws;
  size_t off = (((size_t)n_dst * sizeof(float)) + 255) & ~(size_t)255;
  _Float16* W1T = (_Float16*)(ws + off);
  off += (size_t)HID * ESD * sizeof(_Float16);
  off = (off + 255) & ~(size_t)255;
  _Float16* W2T = (_Float16*)(ws + off);

  {
    int n = out_size > n_dst ? out_size : n_dst;
    zero_kernel<<<(n + 255) / 256, 256, 0, stream>>>(out, out_size, cnt, n_dst);
  }
  {
    int n = HID*ESD + WNUM*HID;
    prep_weights<<<(n + 255) / 256, 256, 0, stream>>>(W1, W2, W1T, W2T);
  }
  edge_kernel<<<(E + ETILE - 1) / ETILE, 32, 0, stream>>>(h_src, edge_vec, edge_scalars,
                                                          b1, b2, src_idx, dst_idx,
                                                          W1T, W2T, out, cnt, E);
  finalize_kernel<<<(out_size + 255) / 256, 256, 0, stream>>>(out, cnt, out_size);
}
